// CrossFramelAttentionBlock_14766097563929
// MI455X (gfx1250) — matrix operand order repaired and checked by reading, not compiled, not run on hardware
//
#include <hip/hip_runtime.h>

// ---------------------------------------------------------------------------
// Types for CDNA5 WMMA
// ---------------------------------------------------------------------------
typedef __attribute__((ext_vector_type(16))) __bf16          v16bf;
typedef __attribute__((ext_vector_type(16))) unsigned short  v16us;
typedef __attribute__((ext_vector_type(8)))  unsigned short  v8us;
typedef __attribute__((ext_vector_type(8)))  float           v8f;
typedef __attribute__((ext_vector_type(4)))  float           v4f;

__device__ __forceinline__ unsigned short f2bf_u(float f) {
  unsigned int u = __builtin_bit_cast(unsigned int, f);
  unsigned int r = u + 0x7FFFu + ((u >> 16) & 1u);   // round-to-nearest-even
  return (unsigned short)(r >> 16);
}
__device__ __forceinline__ float bf2f(unsigned short u) {
  return __builtin_bit_cast(float, (unsigned int)u << 16);
}

// ---------------------------------------------------------------------------
// fp32 -> bf16 bulk convert (8 elems/thread, b128 in / b128 out). n % 8 == 0.
// ---------------------------------------------------------------------------
__global__ __launch_bounds__(256) void cvt_bf16_kernel(
    const float* __restrict__ S, unsigned short* __restrict__ D, int n8)
{
  const int i = blockIdx.x * 256 + threadIdx.x;
  if (i >= n8) return;
  const v4f* s = (const v4f*)(S + (size_t)i * 8);
  const v4f a = s[0], b = s[1];
  v8us o;
  o[0] = f2bf_u(a[0]); o[1] = f2bf_u(a[1]); o[2] = f2bf_u(a[2]); o[3] = f2bf_u(a[3]);
  o[4] = f2bf_u(b[0]); o[5] = f2bf_u(b[1]); o[6] = f2bf_u(b[2]); o[7] = f2bf_u(b[3]);
  *(v8us*)(D + (size_t)i * 8) = o;
}

// ---------------------------------------------------------------------------
// GEMM:  C[M,N] = op( A16[M,K] (bf16,row-major,lda) * W16[N,K]^T (bf16) + bias )
// Output: Cf (fp32; mode 0 store / mode 1 accumulate) or C16 (bf16; mode 0).
// scale = rowscale ? rowscale[m*rs_stride] : 1
// Block 256 thr (8 waves), tile 64(M) x 128(N), K-step 64.
// Each wave: 16x64 -> 4 accumulators; fragments fetched in a batch so the
// 4 WMMAs issue back-to-back behind a single s_wait_dscnt.
// Requires N % 128 == 0, K % 64 == 0.
// ---------------------------------------------------------------------------
__global__ __launch_bounds__(256) void gemm_bf16_kernel(
    const unsigned short* __restrict__ A16, int lda,
    const unsigned short* __restrict__ W16,
    const float* __restrict__ bias,
    float* __restrict__ Cf,
    unsigned short* __restrict__ C16,
    int ldc, int M, int N, int K, int mode,
    const float* __restrict__ rowscale, int rs_stride)
{
  __shared__ unsigned short As[64][72];    //  9.2 KB (pitch 144B, 16B aligned)
  __shared__ unsigned short Ws[128][72];   // 18.4 KB

  const int tid  = threadIdx.x;
  const int lane = tid & 31;
  const int w    = tid >> 5;
  const int wm   = w & 3;        // 4 waves along M
  const int wn   = w >> 2;       // 2 waves along N (64 cols each)
  const int m0   = blockIdx.y * 64;
  const int n0   = blockIdx.x * 128;
  const bool mfull = (m0 + 64 <= M);   // block-uniform: no row guard needed

  v8f acc[4] = {};

  const int aoff = (lane < 16) ? 0 : 8;
  const int boff = (lane < 16) ? 0 : 8;
  const int arow = wm * 16 + (lane & 15);

  for (int k0 = 0; k0 < K; k0 += 64) {
    // ---- stage A tile (64x64) ----
    if (mfull) {
#pragma unroll
      for (int s = tid; s < 512; s += 256) {        // 8 slots/row
        const int r = s >> 3, c8 = (s & 7) * 8;
        *(v8us*)&As[r][c8] =
            *(const v8us*)(A16 + (size_t)(m0 + r) * lda + k0 + c8);
      }
    } else {
#pragma unroll
      for (int s = tid; s < 512; s += 256) {
        const int r = s >> 3, c8 = (s & 7) * 8;
        const int gm = m0 + r;
        v8us v = {};
        if (gm < M) v = *(const v8us*)(A16 + (size_t)gm * lda + k0 + c8);
        *(v8us*)&As[r][c8] = v;
      }
    }
    // ---- stage W tile (128x64) + prefetch next K-tile ----
#pragma unroll
    for (int s = tid; s < 1024; s += 256) {
      const int r = s >> 3, c8 = (s & 7) * 8;
      const unsigned short* p = W16 + (size_t)(n0 + r) * K + k0 + c8;
      *(v8us*)&Ws[r][c8] = *(const v8us*)p;
      if (k0 + 64 < K) __builtin_prefetch(p + 64, 0, 0);  // global_prefetch_b8
    }
    __syncthreads();

#pragma unroll
    for (int d0 = 0; d0 < 64; d0 += 32) {
      // fetch A fragment and ALL 4 B fragments, then issue WMMAs together
      v16us au;
#pragma unroll
      for (int j = 0; j < 8; j++) au[j]     = As[arow][d0 + aoff + j];
#pragma unroll
      for (int j = 0; j < 8; j++) au[8 + j] = As[arow][d0 + 16 + aoff + j];
      v16us bu[4];
#pragma unroll
      for (int nt = 0; nt < 4; nt++) {
        const int cn = wn * 64 + nt * 16 + (lane & 15);
#pragma unroll
        for (int j = 0; j < 16; j++) bu[nt][j] = Ws[cn][d0 + boff + j + ((j >= 8) ? 8 : 0)];
      }
      const v16bf afrag = __builtin_bit_cast(v16bf, au);
#pragma unroll
      for (int nt = 0; nt < 4; nt++) {
        const v16bf bfrag = __builtin_bit_cast(v16bf, bu[nt]);
        acc[nt] = __builtin_amdgcn_wmma_f32_16x16x32_bf16(
            false, afrag, false, bfrag, (short)0, acc[nt], false, false);
      }
    }
    __syncthreads();
  }

  // ---- epilogue (C layout: vgpr r -> row r / r+8 by lane half, col = lane%16)
#pragma unroll
  for (int nt = 0; nt < 4; nt++) {
#pragma unroll
    for (int r = 0; r < 8; r++) {
      const int ml = wm * 16 + r + ((lane >= 16) ? 8 : 0);
      const int gm = m0 + ml;
      if (gm < M) {
        const int gn = n0 + wn * 64 + nt * 16 + (lane & 15);
        float v = acc[nt][r] + bias[gn];
        if (rowscale) v *= rowscale[(size_t)gm * rs_stride];
        const size_t off = (size_t)gm * ldc + gn;
        if (C16)      C16[off] = f2bf_u(v);
        else if (mode) Cf[off] += v;
        else           Cf[off]  = v;
      }
    }
  }
}

// ---------------------------------------------------------------------------
// Attention: one workgroup (128 thr, 4 waves) per (batch nb, head h).
// qkv16 rows = tokens (l*NB+nb), 2304 bf16 cols: q|k|v sections of 768.
// K/V (padded to LKP) staged in LDS (V transposed). Per 16-row q-tile:
// S = Q K^T (WMMA), exact softmax over full row, P -> LDS, O = P V (WMMA).
// O written fp32 [token, 768], scaled by 1/rowsum.
// ---------------------------------------------------------------------------
template <int LKP>
__global__ __launch_bounds__(128) void attn_kernel(
    const unsigned short* __restrict__ qkv, int ldq, int NB,
    int Lq, int Lk,
    float* __restrict__ O, int ldo)
{
  constexpr int NT = LKP / 16;    // score tiles
  constexpr int NC = LKP / 32;    // P@V K-chunks
  constexpr int VP = LKP + 8;     // V^T pitch (16B aligned)

  __shared__ unsigned short Ks[LKP][72];
  __shared__ unsigned short Vts[64][VP];
  __shared__ unsigned short Ps[4][16][LKP];

  const int nb   = blockIdx.x % NB;
  const int h    = blockIdx.x / NB;
  const int tid  = threadIdx.x;
  const int lane = tid & 31;
  const int w    = tid >> 5;

  // ---- stage K rows and V^T (pure bf16 b128 loads) ----
  for (int s = tid; s < LKP * 8; s += 128) {
    const int j = s >> 3, c8 = (s & 7) * 8;
    v8us kv = {}, vv = {};
    if (j < Lk) {
      const unsigned short* base = qkv + (size_t)(j * NB + nb) * ldq + h * 64;
      kv = *(const v8us*)(base + 768 + c8);
      vv = *(const v8us*)(base + 1536 + c8);
    }
    *(v8us*)&Ks[j][c8] = kv;
#pragma unroll
    for (int t = 0; t < 8; t++) Vts[c8 + t][j] = vv[t];
  }
  __syncthreads();

  const int aoff    = (lane < 16) ? 0 : 8;
  const int boff    = (lane < 16) ? 0 : 8;
  const int halfsel = (lane >= 16) ? 8 : 0;

  for (int qt = w; qt * 16 < Lq; qt += 4) {
    v8f S[NT] = {};

    // ---- S = Q K^T (B fragments fetched in pairs, WMMAs issued in pairs) ----
    const int ql = qt * 16 + (lane & 15);
    const bool qvalid = (ql < Lq);
    const unsigned short* qrow =
        qkv + (size_t)((qvalid ? ql : 0) * NB + nb) * ldq + h * 64;
#pragma unroll
    for (int d0 = 0; d0 < 64; d0 += 32) {
      v16us au;
#pragma unroll
      for (int j = 0; j < 8; j++)
        au[j]     = qvalid ? qrow[d0 + aoff + j]      : (unsigned short)0;
#pragma unroll
      for (int j = 0; j < 8; j++)
        au[8 + j] = qvalid ? qrow[d0 + 16 + aoff + j] : (unsigned short)0;
      const v16bf afrag = __builtin_bit_cast(v16bf, au);
#pragma unroll
      for (int nt = 0; nt < NT; nt += 2) {
        v16us bu[2];
#pragma unroll
        for (int u = 0; u < 2; u++) {
          const int cj = (nt + u) * 16 + (lane & 15);
#pragma unroll
          for (int j = 0; j < 16; j++) bu[u][j] = Ks[cj][d0 + boff + j + ((j >= 8) ? 8 : 0)];
        }
#pragma unroll
        for (int u = 0; u < 2; u++) {
          const v16bf bfrag = __builtin_bit_cast(v16bf, bu[u]);
          S[nt + u] = __builtin_amdgcn_wmma_f32_16x16x32_bf16(
              false, afrag, false, bfrag, (short)0, S[nt + u], false, false);
        }
      }
    }

    // ---- scale + mask + exact row softmax (row r+halfsel per lane half) ----
    float ml[8], rs_[8];
#pragma unroll
    for (int r = 0; r < 8; r++) ml[r] = -1e30f;
#pragma unroll
    for (int nt = 0; nt < NT; nt++) {
      const int cj = nt * 16 + (lane & 15);
      const bool valid = (cj < Lk);
#pragma unroll
      for (int r = 0; r < 8; r++) {
        float s = valid ? S[nt][r] * 0.125f : -1e30f;   // 1/sqrt(64)
        S[nt][r] = s;
        ml[r] = fmaxf(ml[r], s);
      }
    }
#pragma unroll
    for (int r = 0; r < 8; r++) {
#pragma unroll
      for (int o = 1; o < 16; o <<= 1) ml[r] = fmaxf(ml[r], __shfl_xor(ml[r], o));
      rs_[r] = 0.f;
    }
#pragma unroll
    for (int nt = 0; nt < NT; nt++) {
#pragma unroll
      for (int r = 0; r < 8; r++) {
        const float p = __expf(S[nt][r] - ml[r]);
        S[nt][r] = p;
        rs_[r] += p;
      }
    }
#pragma unroll
    for (int r = 0; r < 8; r++) {
#pragma unroll
      for (int o = 1; o < 16; o <<= 1) rs_[r] += __shfl_xor(rs_[r], o);
      rs_[r] = 1.0f / rs_[r];
    }

    // ---- P -> per-wave LDS (C layout -> row-major bf16) ----
#pragma unroll
    for (int nt = 0; nt < NT; nt++) {
      const int cj = nt * 16 + (lane & 15);
#pragma unroll
      for (int r = 0; r < 8; r++) Ps[w][r + halfsel][cj] = f2bf_u(S[nt][r]);
    }

    // ---- O = P V (batch 4 B fragments per A fragment) ----
    v8f Oa[4] = {};
    const int pr = lane & 15;
#pragma unroll
    for (int kc = 0; kc < NC; kc++) {
      v16us au;
#pragma unroll
      for (int j = 0; j < 8; j++) au[j]     = Ps[w][pr][kc * 32 + aoff + j];
#pragma unroll
      for (int j = 0; j < 8; j++) au[8 + j] = Ps[w][pr][kc * 32 + 16 + aoff + j];
      v16us bu[4];
#pragma unroll
      for (int dt = 0; dt < 4; dt++) {
        const int dn = dt * 16 + (lane & 15);
#pragma unroll
        for (int j = 0; j < 16; j++) bu[dt][j] = Vts[dn][kc * 32 + boff + j + ((j >= 8) ? 8 : 0)];
      }
      const v16bf afrag = __builtin_bit_cast(v16bf, au);
#pragma unroll
      for (int dt = 0; dt < 4; dt++) {
        const v16bf bfrag = __builtin_bit_cast(v16bf, bu[dt]);
        Oa[dt] = __builtin_amdgcn_wmma_f32_16x16x32_bf16(
            false, afrag, false, bfrag, (short)0, Oa[dt], false, false);
      }
    }

    // ---- store O (normalized, fp32) ----
#pragma unroll
    for (int dt = 0; dt < 4; dt++) {
#pragma unroll
      for (int r = 0; r < 8; r++) {
        const int l = qt * 16 + r + halfsel;
        if (l < Lq) {
          const int dn = dt * 16 + (lane & 15);
          O[(size_t)(l * NB + nb) * ldo + h * 64 + dn] = Oa[dt][r] * rs_[r];
        }
      }
    }
  }
}

// ---------------------------------------------------------------------------
// LayerNorm over D=768, fp32 in -> bf16 out. One wave per row (block = 8 rows).
// ---------------------------------------------------------------------------
__global__ __launch_bounds__(256) void ln_kernel(
    const float* __restrict__ X, const float* __restrict__ g,
    const float* __restrict__ b, unsigned short* __restrict__ Y16, int M)
{
  const int row  = blockIdx.x * 8 + (threadIdx.x >> 5);
  const int lane = threadIdx.x & 31;
  if (row >= M) return;
  const float* x = X + (size_t)row * 768;
  float s = 0.f, s2 = 0.f;
  for (int i = lane; i < 768; i += 32) { const float v = x[i]; s += v; s2 += v * v; }
#pragma unroll
  for (int o = 16; o > 0; o >>= 1) { s += __shfl_xor(s, o); s2 += __shfl_xor(s2, o); }
  const float mean = s * (1.f / 768.f);
  const float var  = s2 * (1.f / 768.f) - mean * mean;
  const float rstd = rsqrtf(var + 1e-5f);
  unsigned short* y = Y16 + (size_t)row * 768;
  for (int i = lane; i < 768; i += 32)
    y[i] = f2bf_u((x[i] - mean) * rstd * g[i] + b[i]);
}

// ---------------------------------------------------------------------------
// Router: R[M,5] = softmax(X16 @ W[5,K]^T + b, axis=-1). One wave per row.
// ---------------------------------------------------------------------------
__global__ __launch_bounds__(256) void router_kernel(
    const unsigned short* __restrict__ X16, int K, const float* __restrict__ W,
    const float* __restrict__ bias, float* __restrict__ R, int M)
{
  const int row  = blockIdx.x * 8 + (threadIdx.x >> 5);
  const int lane = threadIdx.x & 31;
  if (row >= M) return;
  const unsigned short* x = X16 + (size_t)row * K;
  float acc[5] = {0.f, 0.f, 0.f, 0.f, 0.f};
  for (int i = lane; i < K; i += 32) {
    const float v = bf2f(x[i]);
#pragma unroll
    for (int e = 0; e < 5; e++) acc[e] += v * W[(size_t)e * K + i];
  }
#pragma unroll
  for (int e = 0; e < 5; e++)
#pragma unroll
    for (int o = 16; o > 0; o >>= 1) acc[e] += __shfl_xor(acc[e], o);
  if (lane == 0) {
    float z[5], m = -1e30f, s = 0.f;
#pragma unroll
    for (int e = 0; e < 5; e++) { z[e] = acc[e] + bias[e]; m = fmaxf(m, z[e]); }
#pragma unroll
    for (int e = 0; e < 5; e++) { z[e] = __expf(z[e] - m); s += z[e]; }
    const float inv = 1.f / s;
#pragma unroll
    for (int e = 0; e < 5; e++) R[(size_t)row * 5 + e] = z[e] * inv;
  }
}

// ---------------------------------------------------------------------------
// quick-GELU: fp32 in -> bf16 out (8 elems/thread). n % 8 == 0.
// ---------------------------------------------------------------------------
__global__ __launch_bounds__(256) void qgelu_bf16_kernel(
    const float* __restrict__ X, unsigned short* __restrict__ Y16, int n8)
{
  const int i = blockIdx.x * 256 + threadIdx.x;
  if (i >= n8) return;
  const v4f* s = (const v4f*)(X + (size_t)i * 8);
  const v4f a = s[0], b = s[1];
  v8us o;
#pragma unroll
  for (int t = 0; t < 4; t++) {
    const float va = a[t];
    o[t] = f2bf_u(va / (1.f + __expf(-1.702f * va)));
    const float vb = b[t];
    o[4 + t] = f2bf_u(vb / (1.f + __expf(-1.702f * vb)));
  }
  *(v8us*)(Y16 + (size_t)i * 8) = o;
}

// msg0 rows (b*8+t) -> msgP rows (t*4+b)       (fp32)
__global__ __launch_bounds__(256) void permute_bt2tb_kernel(
    const float* __restrict__ src, float* __restrict__ dst)
{
  const int idx = blockIdx.x * 256 + threadIdx.x;
  if (idx >= 32 * 768) return;
  const int row = idx / 768, d = idx % 768;
  const int t = row / 4, b = row % 4;
  dst[(size_t)row * 768 + d] = src[(size_t)(b * 8 + t) * 768 + d];
}

// msgP rows (t*4+b) -> xc rows 6304 + (b*8+t)  (fp32)
__global__ __launch_bounds__(256) void scatter_msg_kernel(
    const float* __restrict__ msgP, float* __restrict__ xc)
{
  const int idx = blockIdx.x * 256 + threadIdx.x;
  if (idx >= 32 * 768) return;
  const int row = idx / 768, d = idx % 768;
  const int b = row / 8, t = row % 8;
  xc[(size_t)(6304 + row) * 768 + d] = msgP[(size_t)(t * 4 + b) * 768 + d];
}

// ---------------------------------------------------------------------------
// Host orchestration
// ---------------------------------------------------------------------------
extern "C" void kernel_launch(void* const* d_in, const int* in_sizes, int n_in,
                              void* d_out, int out_size, void* d_ws, size_t ws_size,
                              hipStream_t stream)
{
  (void)in_sizes; (void)n_in; (void)out_size; (void)ws_size;

  const float* x         = (const float*)d_in[0];
  const float* msg_fc_w  = (const float*)d_in[1];
  const float* msg_fc_b  = (const float*)d_in[2];
  const float* msg_ln_g  = (const float*)d_in[3];
  const float* msg_ln_b  = (const float*)d_in[4];
  const float* msg_wqkv  = (const float*)d_in[5];
  const float* msg_bqkv  = (const float*)d_in[6];
  const float* msg_wo    = (const float*)d_in[7];
  const float* msg_bo    = (const float*)d_in[8];
  const float* attn_wqkv = (const float*)d_in[9];
  const float* attn_bqkv = (const float*)d_in[10];
  const float* attn_wo   = (const float*)d_in[11];
  const float* attn_bo   = (const float*)d_in[12];
  const float* ln1_g     = (const float*)d_in[13];
  const float* ln1_b     = (const float*)d_in[14];
  const float* ln2_g     = (const float*)d_in[15];
  const float* ln2_b     = (const float*)d_in[16];
  const float* cfc_w     = (const float*)d_in[17];
  const float* cfc_b     = (const float*)d_in[18];
  const float* cproj_w   = (const float*)d_in[19];
  const float* cproj_b   = (const float*)d_in[20];
  const float* eh_w      = (const float*)d_in[21];
  const float* eh_b      = (const float*)d_in[22];
  const float* et_w      = (const float*)d_in[23];
  const float* et_b      = (const float*)d_in[24];
  const float* r1_w      = (const float*)d_in[25];
  const float* r1_b      = (const float*)d_in[26];
  const float* r2_w      = (const float*)d_in[27];
  const float* r2_b      = (const float*)d_in[28];

  const int Mfull = 198 * 32;   // 6336 tokens incl. message row
  const int Mx    = 197 * 32;   // 6304 tokens

  // ---- workspace carve-up (256B-aligned regions) ----
  char*  base = (char*)d_ws;
  size_t off  = 0;
  auto alloc = [&](size_t bytes) -> void* {
    void* p = base + off;
    off += (bytes + 255) & ~(size_t)255;
    return p;
  };
  float* xc     = (float*)alloc((size_t)Mfull * 768 * 4);
  float* obuf   = (float*)alloc((size_t)Mfull * 768 * 4);
  float* ohb    = (float*)alloc((size_t)Mx * 3072 * 4);
  float* msg0   = (float*)alloc(32 * 768 * 4);
  float* msgP   = (float*)alloc(32 * 768 * 4);
  float* r1b    = (float*)alloc((size_t)Mx * 5 * 4);
  float* r2b    = (float*)alloc((size_t)Mx * 5 * 4);
  unsigned short* lnb16   = (unsigned short*)alloc((size_t)Mfull * 768 * 2);
  unsigned short* qkvb16  = (unsigned short*)alloc((size_t)Mfull * 2304 * 2);
  unsigned short* obuf16  = (unsigned short*)alloc((size_t)Mfull * 768 * 2);
  unsigned short* ohb16   = (unsigned short*)alloc((size_t)Mx * 3072 * 2);
  unsigned short* x16msg  = (unsigned short*)alloc(32 * 768 * 2);
  unsigned short* wfc16   = (unsigned short*)alloc((size_t)768 * 768 * 2);
  unsigned short* wmq16   = (unsigned short*)alloc((size_t)2304 * 768 * 2);
  unsigned short* wmo16   = (unsigned short*)alloc((size_t)768 * 768 * 2);
  unsigned short* wq16    = (unsigned short*)alloc((size_t)2304 * 768 * 2);
  unsigned short* wo16    = (unsigned short*)alloc((size_t)768 * 768 * 2);
  unsigned short* cfc16   = (unsigned short*)alloc((size_t)3072 * 768 * 2);
  unsigned short* eh16    = (unsigned short*)alloc((size_t)4 * 3072 * 768 * 2);
  unsigned short* cproj16 = (unsigned short*)alloc((size_t)768 * 3072 * 2);
  unsigned short* et16    = (unsigned short*)alloc((size_t)4 * 768 * 3072 * 2);

  const dim3 blk256(256);
  auto cvt = [&](const float* s, unsigned short* d, size_t n) {
    const int n8 = (int)(n / 8);
    cvt_bf16_kernel<<<(n8 + 255) / 256, blk256, 0, stream>>>(s, d, n8);
  };
  auto gemm = [&](const unsigned short* A, int lda, const unsigned short* W,
                  const float* bias, float* Cf, unsigned short* C16, int ldc,
                  int M, int N, int K, int mode, const float* rs, int rss) {
    dim3 grid(N / 128, (M + 63) / 64);
    gemm_bf16_kernel<<<grid, blk256, 0, stream>>>(A, lda, W, bias, Cf, C16, ldc,
                                                  M, N, K, mode, rs, rss);
  };

  // ---- one-time bf16 conversion of all weights + x[0] rows ----
  cvt(msg_fc_w,  wfc16,   (size_t)768 * 768);
  cvt(msg_wqkv,  wmq16,   (size_t)2304 * 768);
  cvt(msg_wo,    wmo16,   (size_t)768 * 768);
  cvt(attn_wqkv, wq16,    (size_t)2304 * 768);
  cvt(attn_wo,   wo16,    (size_t)768 * 768);
  cvt(cfc_w,     cfc16,   (size_t)3072 * 768);
  cvt(eh_w,      eh16,    (size_t)4 * 3072 * 768);
  cvt(cproj_w,   cproj16, (size_t)768 * 3072);
  cvt(et_w,      et16,    (size_t)4 * 768 * 3072);
  cvt(x,         x16msg,  (size_t)32 * 768);

  // --- temporal message branch ---
  gemm(x16msg, 768, wfc16, msg_fc_b, msg0, nullptr, 768, 32, 768, 768, 0, nullptr, 0);
  permute_bt2tb_kernel<<<96, blk256, 0, stream>>>(msg0, msgP);
  ln_kernel<<<4, blk256, 0, stream>>>(msgP, msg_ln_g, msg_ln_b, lnb16, 32);
  gemm(lnb16, 768, wmq16, msg_bqkv, nullptr, qkvb16, 2304, 32, 2304, 768, 0, nullptr, 0);
  attn_kernel<32><<<4 * 12, dim3(128), 0, stream>>>(qkvb16, 2304, 4, 8, 8, obuf, 768);
  cvt(obuf, obuf16, (size_t)32 * 768);
  gemm(obuf16, 768, wmo16, msg_bo, msgP, nullptr, 768, 32, 768, 768, 1, nullptr, 0);

  // --- assemble xc = [x ; msg] ---
  hipMemcpyAsync(xc, x, (size_t)Mx * 768 * sizeof(float),
                 hipMemcpyDeviceToDevice, stream);
  scatter_msg_kernel<<<96, blk256, 0, stream>>>(msgP, xc);

  // --- main attention: xc += MHA(LN1(xc)) ---
  ln_kernel<<<(Mfull + 7) / 8, blk256, 0, stream>>>(xc, ln1_g, ln1_b, lnb16, Mfull);
  gemm(lnb16, 768, wq16, attn_bqkv, nullptr, qkvb16, 2304, Mfull, 2304, 768, 0, nullptr, 0);
  attn_kernel<224><<<32 * 12, dim3(128), 0, stream>>>(qkvb16, 2304, 32, 198, 198, obuf, 768);
  cvt(obuf, obuf16, (size_t)Mfull * 768);
  gemm(obuf16, 768, wo16, attn_bo, xc, nullptr, 768, Mfull, 768, 768, 1, nullptr, 0);

  // --- expert-mixture MLP on x = xc[:197] ---
  ln_kernel<<<(Mx + 7) / 8, blk256, 0, stream>>>(xc, ln2_g, ln2_b, lnb16, Mx);
  router_kernel<<<(Mx + 7) / 8, blk256, 0, stream>>>(lnb16, 768, r1_w, r1_b, r1b, Mx);
  for (int e = 0; e < 5; e++) {
    const unsigned short* Wh = (e == 0) ? cfc16 : eh16 + (size_t)(e - 1) * 3072 * 768;
    const float*          bh = (e == 0) ? cfc_b : eh_b + (size_t)(e - 1) * 3072;
    gemm(lnb16, 768, Wh, bh, ohb, nullptr, 3072, Mx, 3072, 768, e ? 1 : 0, r1b + e, 5);
  }
  const int nOh8 = Mx * 3072 / 8;
  qgelu_bf16_kernel<<<(nOh8 + 255) / 256, blk256, 0, stream>>>(ohb, ohb16, nOh8);
  router_kernel<<<(Mx + 7) / 8, blk256, 0, stream>>>(ohb16, 3072, r2_w, r2_b, r2b, Mx);

  hipMemcpyAsync(d_out, xc, (size_t)Mx * 768 * sizeof(float),
                 hipMemcpyDeviceToDevice, stream);
  for (int e = 0; e < 5; e++) {
    const unsigned short* Wt = (e == 0) ? cproj16 : et16 + (size_t)(e - 1) * 768 * 3072;
    const float*          bt = (e == 0) ? cproj_b : et_b + (size_t)(e - 1) * 768;
    gemm(ohb16, 3072, Wt, bt, (float*)d_out, nullptr, 768, Mx, 768, 3072, 1, r2b + e, 5);
  }
}
